// DecoderAttention_19292993094220
// MI455X (gfx1250) — compile-verified, matrix operand order checked by reading, not run on hardware
//
#include <hip/hip_runtime.h>
#include <hip/hip_bf16.h>

// ---------------------------------------------------------------------------
// DecoderAttention (GRU decoder w/ dot attention, full teacher forcing)
// B=32, S=T=128, E=H=256, V=32000.  Output: (B,T,V) f32 = 524 MB.
// f16 WMMA (v_wmma_f32_16x16x32_f16) for all GEMMs, VALU for attention,
// single-workgroup WMMA recurrence with TDM (tensor_load_to_lds) staging
// of the precomputed input gates, overlapped with the WMMA phase.
// ---------------------------------------------------------------------------

typedef __attribute__((ext_vector_type(16))) _Float16 v16h;
typedef __attribute__((ext_vector_type(8)))  float    v8f;
typedef __attribute__((ext_vector_type(4)))  unsigned u32x4;
typedef __attribute__((ext_vector_type(8)))  unsigned u32x8;

#define WMMA_F16(a, b, c) \
  __builtin_amdgcn_wmma_f32_16x16x32_f16(false, (a), false, (b), (short)0, (c), false, false)

// Load a 16x32 (rows x K) f16 fragment from a row-major matrix (leading dim ld),
// CDNA5 wave32 A-matrix layout (also used for B as B^T rows):
//   lanes 0-15 : row = row0+lane,    elems 0..7 -> K=kb..kb+7,   8..15 -> kb+16..kb+23
//   lanes 16-31: row = row0+lane-16, elems 0..7 -> K=kb+8..kb+15,8..15 -> kb+24..kb+31
__device__ __forceinline__ v16h load_frag_f16(const _Float16* p, int row0, int ld,
                                              int kb, int lane) {
  const int half = lane >> 4;
  const int r    = row0 + (lane & 15);
  const _Float16* q = p + (size_t)r * (size_t)ld + kb + half * 8;
  v16h f;
#pragma unroll
  for (int e = 0; e < 8; ++e) {
    f[e]     = q[e];        // K = kb + half*8 + e
    f[8 + e] = q[16 + e];   // K = kb + 16 + half*8 + e
  }
  return f;
}

// ---------------------------------------------------------------------------
// TDM: 1D contiguous DMA of 24576 f32 (= one GRU step of gi, 32x768) to LDS.
// D# group0: count=1 | lds_addr | global_addr[56:0] | type=2.
// D# group1: data_size=4B, tensor_dim0=tile_dim0=24576, dim1=tile_dim1=1,
//            tensor_dim0_stride=24576.  Not in a cluster -> workgroup_mask=0.
// Issued by one wave only; tracked with TENSORcnt.
// ---------------------------------------------------------------------------
__device__ __forceinline__ void tdm_load_gi_step(const float* gsrc, unsigned lds_off) {
  unsigned long long ga = (unsigned long long)(uintptr_t)gsrc;
  u32x4 g0;
  g0[0] = 1u;                                              // count=1
  g0[1] = lds_off;                                         // lds_addr (bytes)
  g0[2] = (unsigned)(ga & 0xffffffffu);                    // global_addr[31:0]
  g0[3] = (unsigned)((ga >> 32) & 0x01ffffffu) | (2u << 30); // addr[56:32] | type=2
  u32x8 g1;
  g1[0] = 0x00020000u;   // data_size = 2 (4 bytes)
  g1[1] = 0x60000000u;   // tensor_dim0[15:0]=0x6000 in bits 63:48
  g1[2] = 0x00010000u;   // tensor_dim0[31:16]=0 ; tensor_dim1[15:0]=1 (bits 95:80)
  g1[3] = 0x60000000u;   // tensor_dim1[31:16]=0 ; tile_dim0=0x6000 (bits 127:112)
  g1[4] = 1u;            // tile_dim1=1, tile_dim2=0
  g1[5] = 24576u;        // tensor_dim0_stride[31:0]
  g1[6] = 0u;            // stride0[47:32]=0 ; tensor_dim1_stride[15:0]=0
  g1[7] = 0u;
  asm volatile("tensor_load_to_lds %0, %1" :: "s"(g0), "s"(g1) : "memory");
}

// ---------------------------------------------------------------------------
// Kernel 0: f32 -> f16 convert
// ---------------------------------------------------------------------------
__global__ __launch_bounds__(256) void cvt_f16_kernel(const float* __restrict__ src,
                                                      _Float16* __restrict__ dst, int n) {
  int i = blockIdx.x * 256 + threadIdx.x;
  if (i < n) dst[i] = (_Float16)src[i];
}

// ---------------------------------------------------------------------------
// Kernel 1: embedding gather + dot attention + softmax + context.
// Writes x_seq f16, row = t*32 + b, 512 cols = [emb(256) | context(256)].
// ---------------------------------------------------------------------------
__global__ __launch_bounds__(256) void attn_kernel(const float* __restrict__ enc,
                                                   const int* __restrict__ target,
                                                   const float* __restrict__ embedding,
                                                   _Float16* __restrict__ x16) {
  __shared__ float embrow[256];
  __shared__ float sc[128];
  __shared__ float red[128];
  const int b   = blockIdx.x;
  const int tid = threadIdx.x;
  const float* encB = enc + (size_t)b * 128 * 256;

  for (int tt = 0; tt < 16; ++tt) {
    const int t = blockIdx.y * 16 + tt;
    const int token = (t == 0) ? 0 : target[b * 128 + (t - 1)];   // SOS = 0
    embrow[tid] = embedding[(size_t)token * 256 + tid];
    __syncthreads();

    if (tid < 128) {                    // scores[s] = emb . enc[b,s,:]
      const float* er = encB + (size_t)tid * 256;
      float s = 0.f;
#pragma unroll 8
      for (int e = 0; e < 256; ++e) s = fmaf(embrow[e], er[e], s);
      sc[tid] = s;
      red[tid] = s;
    }
    __syncthreads();
    for (int off = 64; off > 0; off >>= 1) {
      if (tid < off) red[tid] = fmaxf(red[tid], red[tid + off]);
      __syncthreads();
    }
    const float mx = red[0];
    __syncthreads();
    if (tid < 128) {
      float e = expf(sc[tid] - mx);
      sc[tid] = e;
      red[tid] = e;
    }
    __syncthreads();
    for (int off = 64; off > 0; off >>= 1) {
      if (tid < off) red[tid] += red[tid + off];
      __syncthreads();
    }
    const float inv = 1.0f / red[0];
    __syncthreads();

    float ctx = 0.f;
#pragma unroll 4
    for (int s = 0; s < 128; ++s) ctx = fmaf(sc[s], encB[(size_t)s * 256 + tid], ctx);
    ctx *= inv;

    const size_t row = (size_t)(t * 32 + b) * 512;
    x16[row + tid]       = (_Float16)embrow[tid];
    x16[row + 256 + tid] = (_Float16)ctx;
    __syncthreads();
  }
}

// ---------------------------------------------------------------------------
// Kernel 2: gi = x_seq(4096x512) @ W_ih^T (768 rows) + b_ih -> f32 (4096x768)
// Each wave: 16(M) x 64(N).  3072 waves = 384 blocks x 8 waves.
// ---------------------------------------------------------------------------
__global__ __launch_bounds__(256) void gi_gemm_kernel(const _Float16* __restrict__ X,
                                                      const _Float16* __restrict__ Wih,
                                                      const float* __restrict__ b_ih,
                                                      float* __restrict__ gi) {
  const int gw   = (blockIdx.x * 256 + threadIdx.x) >> 5;
  const int lane = threadIdx.x & 31;
  const int mt = gw / 12;
  const int ns = gw % 12;

  v8f acc[4];
#pragma unroll
  for (int s = 0; s < 4; ++s)
#pragma unroll
    for (int g = 0; g < 8; ++g) acc[s][g] = 0.f;

  for (int kb = 0; kb < 512; kb += 32) {
    v16h a = load_frag_f16(X, mt * 16, 512, kb, lane);
#pragma unroll
    for (int s = 0; s < 4; ++s) {
      v16h bfr = load_frag_f16(Wih, ns * 64 + s * 16, 512, kb, lane);
      acc[s] = WMMA_F16(a, bfr, acc[s]);
    }
  }
#pragma unroll
  for (int s = 0; s < 4; ++s)
#pragma unroll
    for (int g = 0; g < 8; ++g) {
      const int m = mt * 16 + ((lane < 16) ? g : 8 + g);
      const int n = ns * 64 + s * 16 + (lane & 15);
      gi[(size_t)m * 768 + n] = acc[s][g] + b_ih[n];
    }
}

// ---------------------------------------------------------------------------
// Kernel 3: sequential GRU.  One workgroup, 512 threads (16 waves).
// Per step: TDM stages gi[t] (96KB) into LDS overlapped with the WMMA phase
// computing gh = h(32x256) @ W_hh^T (96 tiles, 6/wave); then VALU gates.
// LDS: gh16 48KB + h16 16KB + gi stage 96KB = 160KB.
// ---------------------------------------------------------------------------
__global__ __launch_bounds__(512) void gru_kernel(const float* __restrict__ h0,
                                                  const float* __restrict__ gi,
                                                  const _Float16* __restrict__ Whh,
                                                  const float* __restrict__ b_hh,
                                                  _Float16* __restrict__ hs16) {
  __shared__ _Float16 gh16[32 * 768];     // 48 KB
  __shared__ _Float16 h16[32 * 256];      // 16 KB
  __shared__ float    gilds[32 * 768];    // 96 KB TDM staging
  const int tid  = threadIdx.x;
  const int lane = tid & 31;
  const int wid  = tid >> 5;
  const unsigned gilds_off = (unsigned)(uintptr_t)&gilds[0];   // LDS byte offset

  for (int q = tid; q < 32 * 256; q += 512) h16[q] = (_Float16)h0[q];
  __syncthreads();

  for (int t = 0; t < 128; ++t) {
    // kick off DMA of gi rows [t*32 .. t*32+31] while WMMA runs (wave 0 only)
    if (wid == 0) tdm_load_gi_step(gi + (size_t)t * 32 * 768, gilds_off);

    // ---- gh = h @ W_hh^T : 2 M-tiles x 48 N-tiles, K=256 ----
#pragma unroll 2
    for (int i = 0; i < 6; ++i) {
      const int tile = wid * 6 + i;       // 0..95, uniform per wave
      const int mtv  = tile & 1;
      const int ntv  = tile >> 1;
      v8f acc;
#pragma unroll
      for (int g = 0; g < 8; ++g) acc[g] = 0.f;
      for (int kb = 0; kb < 256; kb += 32) {
        v16h a   = load_frag_f16(&h16[0], mtv * 16, 256, kb, lane);
        v16h bfr = load_frag_f16(Whh, ntv * 16, 256, kb, lane);
        acc = WMMA_F16(a, bfr, acc);
      }
#pragma unroll
      for (int g = 0; g < 8; ++g) {
        const int m = mtv * 16 + ((lane < 16) ? g : 8 + g);
        const int n = ntv * 16 + (lane & 15);
        gh16[m * 768 + n] = (_Float16)acc[g];
      }
    }
    if (wid == 0) __builtin_amdgcn_s_wait_tensorcnt(0);   // gi[t] landed in LDS
    __syncthreads();                                      // gh16 + gilds visible

    // ---- gates + state update (all operands now in LDS) ----
    for (int q = tid; q < 32 * 256; q += 512) {
      const int b = q >> 8;
      const int j = q & 255;
      const int row = b * 768;
      const float gr = gilds[row + j]       + (float)gh16[row + j]       + b_hh[j];
      const float gz = gilds[row + 256 + j] + (float)gh16[row + 256 + j] + b_hh[256 + j];
      const float hn = (float)gh16[row + 512 + j] + b_hh[512 + j];
      const float gn = gilds[row + 512 + j];
      const float r  = 1.f / (1.f + expf(-gr));
      const float z  = 1.f / (1.f + expf(-gz));
      const float n  = tanhf(gn + r * hn);
      const float hp = (float)h16[q];
      const float hv = (1.f - z) * n + z * hp;
      h16[q] = (_Float16)hv;
      hs16[(size_t)(t * 32 + b) * 256 + j] = (_Float16)hv;
    }
    __syncthreads();   // protects gilds/h16 for next iteration's DMA + WMMA
  }
}

// ---------------------------------------------------------------------------
// Kernel 4: logits = hs(4096x256) @ W_out^T (32000 rows) + b_out -> d_out f32.
// Each wave: 32(M) x 64(N) strip (B fragments reused across 2 M-tiles).
// 64000 waves = 8000 blocks x 8 waves.  Row remap m=t*32+b -> out[b,t,:].
// ---------------------------------------------------------------------------
__global__ __launch_bounds__(256) void logits_kernel(const _Float16* __restrict__ H,
                                                     const _Float16* __restrict__ Wout,
                                                     const float* __restrict__ b_out,
                                                     float* __restrict__ out) {
  const int gw   = (blockIdx.x * 256 + threadIdx.x) >> 5;  // 0..63999
  const int lane = threadIdx.x & 31;
  const int mt = gw / 500;       // 0..127 (M strips of 32 rows)
  const int ns = gw % 500;       // 0..499 (N strips of 64 cols)

  v8f acc[2][4];
#pragma unroll
  for (int r = 0; r < 2; ++r)
#pragma unroll
    for (int s = 0; s < 4; ++s)
#pragma unroll
      for (int g = 0; g < 8; ++g) acc[r][s][g] = 0.f;

  for (int kb = 0; kb < 256; kb += 32) {
    v16h a0 = load_frag_f16(H, mt * 32,      256, kb, lane);
    v16h a1 = load_frag_f16(H, mt * 32 + 16, 256, kb, lane);
#pragma unroll
    for (int s = 0; s < 4; ++s) {
      v16h bfr = load_frag_f16(Wout, ns * 64 + s * 16, 256, kb, lane);
      acc[0][s] = WMMA_F16(a0, bfr, acc[0][s]);
      acc[1][s] = WMMA_F16(a1, bfr, acc[1][s]);
    }
  }
#pragma unroll
  for (int r = 0; r < 2; ++r)
#pragma unroll
    for (int s = 0; s < 4; ++s)
#pragma unroll
      for (int g = 0; g < 8; ++g) {
        const int m  = mt * 32 + r * 16 + ((lane < 16) ? g : 8 + g);
        const int n  = ns * 64 + s * 16 + (lane & 15);
        const int bb = m & 31;          // batch
        const int tt = m >> 5;          // time
        out[((size_t)bb * 128 + tt) * 32000 + n] = acc[r][s][g] + b_out[n];
      }
}

// ---------------------------------------------------------------------------
// Host launcher
// ---------------------------------------------------------------------------
extern "C" void kernel_launch(void* const* d_in, const int* in_sizes, int n_in,
                              void* d_out, int out_size, void* d_ws, size_t ws_size,
                              hipStream_t stream) {
  (void)in_sizes; (void)n_in; (void)out_size; (void)ws_size;

  const float* enc       = (const float*)d_in[0];   // (32,128,256)
  const float* enc_h     = (const float*)d_in[1];   // (1,32,256)
  const int*   target    = (const int*)  d_in[2];   // (32,128)
  const float* embedding = (const float*)d_in[3];   // (32000,256)
  const float* W_ih      = (const float*)d_in[4];   // (768,512)
  const float* W_hh      = (const float*)d_in[5];   // (768,256)
  const float* b_ih      = (const float*)d_in[6];   // (768,)
  const float* b_hh      = (const float*)d_in[7];   // (768,)
  const float* W_out     = (const float*)d_in[8];   // (32000,256)
  const float* b_out     = (const float*)d_in[9];   // (32000,)
  float*       out       = (float*)d_out;           // (32,128,32000)

  // workspace carve (all sizes 256B-multiples)
  char* w = (char*)d_ws;
  _Float16* Wih16  = (_Float16*)(w);                        // 768*512*2   =   786432
  _Float16* Whh16  = (_Float16*)(w + 786432);               // 768*256*2   =   393216
  _Float16* Wout16 = (_Float16*)(w + 1179648);              // 32000*256*2 = 16384000
  _Float16* x16    = (_Float16*)(w + 17563648);             // 4096*512*2  =  4194304
  float*    gi     = (float*)   (w + 21757952);             // 4096*768*4  = 12582912
  _Float16* hs16   = (_Float16*)(w + 34340864);             // 4096*256*2  =  2097152
  // total = 36438016 bytes (~34.8 MB)

  // 0) weight converts
  cvt_f16_kernel<<<(768 * 512 + 255) / 256, 256, 0, stream>>>(W_ih, Wih16, 768 * 512);
  cvt_f16_kernel<<<(768 * 256 + 255) / 256, 256, 0, stream>>>(W_hh, Whh16, 768 * 256);
  cvt_f16_kernel<<<(32000 * 256 + 255) / 256, 256, 0, stream>>>(W_out, Wout16, 32000 * 256);

  // 1) embedding + attention -> x_seq f16 (row = t*32+b)
  attn_kernel<<<dim3(32, 8), 256, 0, stream>>>(enc, target, embedding, x16);

  // 2) gi = x_seq @ W_ih^T + b_ih   (4096x512x768)
  gi_gemm_kernel<<<384, 256, 0, stream>>>(x16, Wih16, b_ih, gi);

  // 3) sequential GRU (single workgroup, WMMA + TDM-staged gi)
  gru_kernel<<<1, 512, 0, stream>>>(enc_h, gi, Whh16, b_hh, hs16);

  // 4) logits = hs @ W_out^T + b_out  (4096x256x32000)
  logits_kernel<<<8000, 256, 0, stream>>>(hs16, Wout16, b_out, out);
}
